// SynthesisLayer_17334488006827
// MI455X (gfx1250) — matrix operand order repaired and checked by reading, not compiled, not run on hardware
//
#include <hip/hip_runtime.h>
#include <hip/hip_bf16.h>
#include <stdint.h>

// ---------------- problem constants ----------------
#define BB    8
#define CIN   256
#define COUT  256
#define HH    64
#define WW    64
#define WDIM  512
#define PH    66            // padded height (1 halo each side)
#define PW    80            // padded width  (1+1 halo, padded to 80 for 16B chunks)

// ---------------- workspace layout (bytes) ----------------
#define OFF_S       0u          // 2048 f32   (style s[b][c])
#define OFF_W2      8192u       // 65536 f32  (w2[o][i])
#define OFF_DEMOD   270336u     // 2048 f32   (demod[b][o])
#define OFF_PACKA   278528u     // 589824 bf16 = 1179648 B (WMMA A fragments)
#define OFF_XM      1458176u    // 8*66*80*256 bf16 = 21626880 B (padded NHWC xm)

typedef __bf16 v16bf __attribute__((ext_vector_type(16)));
typedef float  v8f   __attribute__((ext_vector_type(8)));

// ---------------- async copy helpers (guarded) ----------------
#if defined(__has_builtin)
#  if __has_builtin(__builtin_amdgcn_global_load_async_to_lds_b128)
#    define HAVE_ASYNC 1
#  endif
#endif
#ifndef HAVE_ASYNC
#  define HAVE_ASYNC 0
#endif

#if HAVE_ASYNC
typedef int v4i_vs __attribute__((vector_size(4 * sizeof(int))));
typedef __attribute__((address_space(1))) v4i_vs gv4i;
typedef __attribute__((address_space(3))) v4i_vs lv4i;
#endif

__device__ __forceinline__ void async_copy16(const char* g, char* l) {
#if HAVE_ASYNC
  __builtin_amdgcn_global_load_async_to_lds_b128((gv4i*)g, (lv4i*)l, 0, 0);
#else
  *(uint4*)l = *(const uint4*)g;
#endif
}

__device__ __forceinline__ void wait_async() {
#if HAVE_ASYNC
#  if __has_builtin(__builtin_amdgcn_s_wait_asynccnt)
  __builtin_amdgcn_s_wait_asynccnt(0);
#  else
  asm volatile("s_wait_asynccnt 0" ::: "memory");
#  endif
#endif
}

// ================= precompute kernels =================

// s[b][c] = dot(w[b,:], style_weight[c,:]) / sqrt(512) + style_bias[c]
__global__ __launch_bounds__(256) void k_style(const float* __restrict__ wv,
                                               const float* __restrict__ sw,
                                               const float* __restrict__ sb,
                                               float* __restrict__ s) {
  int b = blockIdx.x, c = threadIdx.x;
  float acc = 0.f;
  for (int d = 0; d < WDIM; ++d) acc += wv[b * WDIM + d] * sw[c * WDIM + d];
  s[b * CIN + c] = acc * 0.04419417382415922f + sb[c];   // 1/sqrt(512)
}

// w2[o][i] = (1/2304) * sum_t cw[o][i][t]^2
__global__ __launch_bounds__(256) void k_w2(const float* __restrict__ cw,
                                            float* __restrict__ w2) {
  int o = blockIdx.x, i = threadIdx.x;
  const float* p = cw + ((size_t)o * CIN + i) * 9;
  float a = 0.f;
#pragma unroll
  for (int t = 0; t < 9; ++t) a += p[t] * p[t];
  w2[o * CIN + i] = a * (1.0f / 2304.0f);
}

// demod[b][o] = rsqrt(sum_i w2[o][i] * s[b][i]^2 + eps)
__global__ __launch_bounds__(256) void k_demod(const float* __restrict__ s,
                                               const float* __restrict__ w2,
                                               float* __restrict__ dm) {
  int b = blockIdx.x, o = threadIdx.x;
  float a = 0.f;
  for (int i = 0; i < CIN; ++i) {
    float sv = s[b * CIN + i];
    a += w2[o * CIN + i] * sv * sv;
  }
  dm[b * COUT + o] = rsqrtf(a + 1e-8f);
}

// Pack conv weights (scaled 1/48) into bf16, in a contiguous-k 16-bit
// A-matrix (16x32) lane/VGPR layout so the main loop loads 32B/lane.
// packA[oT][cb][t][lane][j], lane: m=lane&15, khalf=lane>>4,
// K = khalf*16 + j
__global__ __launch_bounds__(256) void k_pack(const float* __restrict__ cw,
                                              __bf16* __restrict__ pA) {
  int idx  = blockIdx.x * 256 + threadIdx.x;      // < 589824
  int j    = idx & 15;
  int lane = (idx >> 4) & 31;
  int t    = (idx >> 9) % 9;
  int rest = idx / (512 * 9);
  int cb   = rest & 7;
  int oT   = rest >> 3;
  int o    = oT * 16 + (lane & 15);
  int kh2  = lane >> 4;
  int kidx = kh2 * 16 + j;
  int cin  = cb * 32 + kidx;
  float v  = cw[((size_t)o * CIN + cin) * 9 + t] * 0.020833333333333332f; // 1/sqrt(2304)
  pA[idx] = (__bf16)v;
}

// Modulated input, padded NHWC bf16: xm[b][r(66)][col(80)][c(256)]
__global__ __launch_bounds__(256) void k_xm(const float* __restrict__ x,
                                            const float* __restrict__ s,
                                            __bf16* __restrict__ xm) {
  size_t idx = (size_t)blockIdx.x * 256 + threadIdx.x;  // < 8*66*80*256
  int c   = (int)(idx & 255);
  int col = (int)((idx >> 8) % PW);
  size_t r2 = idx / (PW * 256);
  int r = (int)(r2 % PH);
  int b = (int)(r2 / PH);
  float v = 0.f;
  if (r >= 1 && r <= HH && col >= 1 && col <= WW)
    v = x[(((size_t)b * CIN + c) * HH + (r - 1)) * WW + (col - 1)] * s[b * CIN + c];
  xm[idx] = (__bf16)v;
}

// ================= main implicit-GEMM WMMA conv =================
// grid = (H=64, COUT/32=8, B=8), block = 256 (8 waves).
// wave -> (cout tile 0/1, pixel tile 0..3). K loop: 8 cin-blocks x 9 taps.
__global__ __launch_bounds__(256) void k_conv(const float* __restrict__ noise,
                                              const float* __restrict__ nw,
                                              const float* __restrict__ bias,
                                              const char*  __restrict__ ws,
                                              float* __restrict__ out) {
  __shared__ __align__(32) char lbuf[2][15360];  // 32cin x 3rows x 80cols bf16, x2

  const int h    = blockIdx.x;
  const int oG   = blockIdx.y;
  const int b    = blockIdx.z;
  const int tid  = threadIdx.x;
  const int lane = tid & 31;
  const int wv   = tid >> 5;
  const int oT   = oG * 2 + (wv >> 2);   // 0..15
  const int pt   = wv & 3;               // pixel tile
  const int o16  = oT * 16;

  const char*   xmB   = ws + OFF_XM;
  const __bf16* packA = (const __bf16*)(ws + OFF_PACKA);
  const float*  demod = (const float*)(ws + OFF_DEMOD);

  // stage one cin-block (32 ch x 3 rows x 80 cols bf16) into lbuf[bufi]
  auto stage = [&](int cb, int bufi) {
    for (int c = tid; c < 960; c += 256) {          // 960 x 16B chunks
      int r   = c / 320;
      int rem = c - r * 320;
      int col = rem >> 2;
      int q   = rem & 3;
      const char* g = xmB +
          ((((size_t)(b * PH + h + r)) * PW + col) * CIN + (size_t)cb * 32) * 2 + q * 16;
      char* l = &lbuf[bufi][(r * PW + col) * 64 + q * 16];
      async_copy16(g, l);
    }
  };

  v8f acc = {};
  const int n   = lane & 15;
  const int kh2 = lane >> 4;

  stage(0, 0);
  wait_async();
  __syncthreads();

  for (int cb = 0; cb < 8; ++cb) {
    int cur = cb & 1;
    if (cb < 7) stage(cb + 1, cur ^ 1);

    const __bf16* aP = packA + ((size_t)oT * 8 + cb) * 9 * 512 + lane * 16;
#pragma unroll
    for (int t = 0; t < 9; ++t) {
      int r  = t / 3;
      int kw = t - r * 3;
      v16bf a = *(const v16bf*)(aP + (size_t)t * 512);
      int col = pt * 16 + n + kw;                       // padded col of (w + kw - 1)
      const char* lp = &lbuf[cur][(r * PW + col) * 64 + kh2 * 32];
      v16bf bm = *(const v16bf*)lp;
      acc = __builtin_amdgcn_wmma_f32_16x16x32_bf16(
          false, a, false, bm, (short)0, acc, false, false);
    }

    wait_async();
    __syncthreads();
  }

  // epilogue: demod, noise, bias, LeakyReLU(0.2)
  const int wpix = pt * 16 + n;
  const float nz = noise[((size_t)b * HH + h) * WW + wpix];
#pragma unroll
  for (int r = 0; r < 8; ++r) {
    int o   = o16 + r + 8 * kh2;                        // C layout: M = r + 8*(lane>>4)
    float y = acc[r] * demod[b * COUT + o];
    y += nw[o] * nz + bias[o];
    y = (y >= 0.f) ? y : 0.2f * y;
    out[(((size_t)b * COUT + o) * HH + h) * WW + wpix] = y;
  }
}

// ================= launcher =================
extern "C" void kernel_launch(void* const* d_in, const int* in_sizes, int n_in,
                              void* d_out, int out_size, void* d_ws, size_t ws_size,
                              hipStream_t stream) {
  const float* x     = (const float*)d_in[0];
  const float* wvec  = (const float*)d_in[1];
  const float* noise = (const float*)d_in[2];
  const float* sw    = (const float*)d_in[3];
  const float* sb    = (const float*)d_in[4];
  const float* cw    = (const float*)d_in[5];
  const float* nw    = (const float*)d_in[6];
  const float* bias  = (const float*)d_in[7];
  float* out = (float*)d_out;
  char*  ws  = (char*)d_ws;

  float*  s_  = (float*)(ws + OFF_S);
  float*  w2_ = (float*)(ws + OFF_W2);
  float*  dm_ = (float*)(ws + OFF_DEMOD);
  __bf16* pA_ = (__bf16*)(ws + OFF_PACKA);
  __bf16* xm_ = (__bf16*)(ws + OFF_XM);

  k_style<<<8,    256, 0, stream>>>(wvec, sw, sb, s_);
  k_w2   <<<256,  256, 0, stream>>>(cw, w2_);
  k_demod<<<8,    256, 0, stream>>>(s_, w2_, dm_);
  k_pack <<<2304, 256, 0, stream>>>(cw, pA_);
  k_xm   <<<42240,256, 0, stream>>>(x, s_, xm_);

  dim3 grid(HH, COUT / 32, BB);
  k_conv<<<grid, 256, 0, stream>>>(noise, nw, bias, (const char*)ws, out);
}
